// Classifier_8856222564744
// MI455X (gfx1250) — matrix operand order repaired and checked by reading, not compiled, not run on hardware
//
#include <hip/hip_runtime.h>
#include <hip/hip_bf16.h>

// ---- WMMA vector types (gfx1250, wave32) ----
typedef __attribute__((ext_vector_type(16))) _Float16 v16h;
typedef __attribute__((ext_vector_type(8)))  _Float16 v8h;
typedef __attribute__((ext_vector_type(4)))  _Float16 v4h;
typedef __attribute__((ext_vector_type(8)))  float    v8f;

#define WAVES            8
#define EDGES_PER_WAVE   16
#define EDGES_PER_BLOCK  (WAVES * EDGES_PER_WAVE)   // 128
#define EF1_STRIDE       232                         // halfs per row (464B, 8/16B aligned, conflict-friendly)

constexpr int   Dfeat = 100;   // per-node feature dim
constexpr float NEG   = 0.01f; // leaky_relu slope

__device__ __forceinline__ float lrelu(float x) { return x >= 0.f ? x : NEG * x; }

// ---------------------------------------------------------------------------
// Pre-pass: convert W1/W2/W3 to f16 and swizzle into WMMA B-fragment layout.
// Fragment = 32 lanes x 16 halfs (512 halfs). Lane l holds column N = n*16+(l&15),
// K values Kbase..Kbase+15 contiguous, Kbase = s*32 + (l>>4)*16.
// W1 K-dim is padded 200 -> 224: padded k<100 -> drug row k; 100..111 -> 0;
// 112..211 -> disease row (k-112) i.e. source row k-12 of concat W1; 212..223 -> 0.
// ---------------------------------------------------------------------------
__global__ void pack_weights(const float* __restrict__ W1, const float* __restrict__ W2,
                             const float* __restrict__ W3,
                             _Float16* __restrict__ w1p, _Float16* __restrict__ w2p,
                             _Float16* __restrict__ w3p) {
  const int f = blockIdx.x;                  // fragment id: 0..55 W1, 56..63 W2, 64 W3
  for (int p = threadIdx.x; p < 512; p += blockDim.x) {
    const int lane  = p >> 4;
    const int j     = p & 15;
    const int nloc  = lane & 15;
    const int khalf = lane >> 4;
    if (f < 56) {
      const int s = f >> 3, n = f & 7;
      const int N = n * 16 + nloc;
      const int k = s * 32 + khalf * 16 + j; // padded K in [0,224)
      float v;
      if (k < 100)      v = W1[k * 128 + N];
      else if (k < 112) v = 0.f;
      else if (k < 212) v = W1[(k - 12) * 128 + N];
      else              v = 0.f;
      w1p[f * 512 + p] = (_Float16)v;
    } else if (f < 64) {
      const int g = f - 56, s = g >> 1, n = g & 1;
      const int N = n * 16 + nloc;
      const int k = s * 32 + khalf * 16 + j; // [0,128)
      w2p[g * 512 + p] = (_Float16)W2[k * 32 + N];
    } else {
      const int N = nloc;
      const int k = khalf * 16 + j;          // [0,32)
      w3p[p] = (_Float16)W3[k * 16 + N];
    }
  }
}

// ---------------------------------------------------------------------------
// Main fused kernel: gather -> (f16 WMMA) 200->128->32->16 -> dot -> out
// Each wave processes one 16-edge M-tile; 8 waves per block.
// ---------------------------------------------------------------------------
__global__ __launch_bounds__(256) void mlp_edge_kernel(
    const float* __restrict__ xd, const float* __restrict__ xs,
    const int*   __restrict__ eli,
    const _Float16* __restrict__ w1p, const _Float16* __restrict__ w2p,
    const _Float16* __restrict__ w3p,
    const float* __restrict__ b1, const float* __restrict__ b2,
    const float* __restrict__ b3, const float* __restrict__ W4,
    const float* __restrict__ b4,
    float* __restrict__ out, int E)
{
  __shared__ __align__(16) _Float16 lds[WAVES * 16 * EF1_STRIDE]; // 59,392 B

  const int tid    = threadIdx.x;
  const int wave   = tid >> 5;
  const int lane   = tid & 31;
  const int laneM  = lane & 15;
  const int laneHi = lane >> 4;

  _Float16* wbase = lds + wave * 16 * EF1_STRIDE;

  const int e0 = (blockIdx.x * WAVES + wave) * EDGES_PER_WAVE;
  const int eClamped = min(e0 + laneM, E - 1);
  const int srcI = eli[eClamped];       // edge_label_index[0][e]
  const int dstI = eli[E + eClamped];   // edge_label_index[1][e]

  // ---- Phase 1: stage ef tile [16 x 224] f16 (concat + zero pad) into LDS ----
  // 100 features = 25 x float4; lanes 0..24 load float4 from both tables and
  // store packed f16 b64; lanes 25..27 write the 12-half zero pad per region.
  #pragma unroll
  for (int r = 0; r < 16; ++r) {
    const int si = __shfl(srcI, r);
    const int di = __shfl(dstI, r);
    _Float16* row = wbase + r * EF1_STRIDE;
    if (lane < 25) {
      const float4 vs = *(const float4*)(xd + (long)si * Dfeat + 4 * lane);
      const float4 vd = *(const float4*)(xs + (long)di * Dfeat + 4 * lane);
      v4h hs = { (_Float16)vs.x, (_Float16)vs.y, (_Float16)vs.z, (_Float16)vs.w };
      v4h hd = { (_Float16)vd.x, (_Float16)vd.y, (_Float16)vd.z, (_Float16)vd.w };
      *(v4h*)(row + 4 * lane)       = hs;   // drug features [0,100)
      *(v4h*)(row + 112 + 4 * lane) = hd;   // disease features at [112,212)
    } else if (lane < 28) {
      v4h z = {};
      *(v4h*)(row + 4 * lane)       = z;    // pad [100,112)
      *(v4h*)(row + 112 + 4 * lane) = z;    // pad [212,224)
    }
  }
  __syncthreads();

  // ---- Phase 2: load all layer-1 A fragments (K=224 -> 7 k-steps) ----
  v16h a1[7];
  {
    const _Float16* row = wbase + laneM * EF1_STRIDE;
    #pragma unroll
    for (int s = 0; s < 7; ++s) {
      const int k0 = s * 32 + laneHi * 16;
      v8h lo = *(const v8h*)(row + k0);
      v8h hi = *(const v8h*)(row + k0 + 8);
      a1[s] = __builtin_shufflevector(lo, hi, 0,1,2,3,4,5,6,7,8,9,10,11,12,13,14,15);
    }
  }
  __syncthreads();

  // ---- Phase 3: layer 1 — [16x224] x [224x128], 56 WMMAs; write ef2 f16 [16x128] ----
  _Float16* ef2 = wbase;
  #pragma unroll
  for (int n = 0; n < 8; ++n) {
    v8f c = {};
    #pragma unroll
    for (int s = 0; s < 7; ++s) {
      v16h b = *(const v16h*)(w1p + (((s * 8 + n) * 32) + lane) * 16);
      c = __builtin_amdgcn_wmma_f32_16x16x32_f16(false, a1[s], false, b, (short)0, c, false, false);
    }
    const float bb = b1[n * 16 + laneM];
    #pragma unroll
    for (int r = 0; r < 8; ++r) {
      ef2[(r + laneHi * 8) * 128 + n * 16 + laneM] = (_Float16)lrelu(c[r] + bb);
    }
  }
  __syncthreads();

  // ---- Phase 4: layer-2 A fragments (K=128 -> 4 k-steps) ----
  v16h a2[4];
  {
    const _Float16* row = wbase + laneM * 128;
    #pragma unroll
    for (int s = 0; s < 4; ++s) {
      const int k0 = s * 32 + laneHi * 16;
      v8h lo = *(const v8h*)(row + k0);
      v8h hi = *(const v8h*)(row + k0 + 8);
      a2[s] = __builtin_shufflevector(lo, hi, 0,1,2,3,4,5,6,7,8,9,10,11,12,13,14,15);
    }
  }
  __syncthreads();

  // ---- Phase 5: layer 2 — [16x128] x [128x32], 8 WMMAs; write ef3 f16 [16x32] ----
  _Float16* ef3 = wbase;
  #pragma unroll
  for (int n = 0; n < 2; ++n) {
    v8f c = {};
    #pragma unroll
    for (int s = 0; s < 4; ++s) {
      v16h b = *(const v16h*)(w2p + (((s * 2 + n) * 32) + lane) * 16);
      c = __builtin_amdgcn_wmma_f32_16x16x32_f16(false, a2[s], false, b, (short)0, c, false, false);
    }
    const float bb = b2[n * 16 + laneM];
    #pragma unroll
    for (int r = 0; r < 8; ++r) {
      ef3[(r + laneHi * 8) * 32 + n * 16 + laneM] = (_Float16)lrelu(c[r] + bb);
    }
  }
  __syncthreads();

  // ---- Phase 6: layer-3 A fragment (K=32 -> 1 k-step) ----
  v16h a3;
  {
    const _Float16* row = wbase + laneM * 32;
    const int k0 = laneHi * 16;
    v8h lo = *(const v8h*)(row + k0);
    v8h hi = *(const v8h*)(row + k0 + 8);
    a3 = __builtin_shufflevector(lo, hi, 0,1,2,3,4,5,6,7,8,9,10,11,12,13,14,15);
  }
  __syncthreads();

  // ---- Phase 7: layer 3 — [16x32] x [32x16], 1 WMMA; write f32 [16x16] ----
  float* fbuf = (float*)wbase;
  {
    v8f c = {};
    v16h b = *(const v16h*)(w3p + lane * 16);
    c = __builtin_amdgcn_wmma_f32_16x16x32_f16(false, a3, false, b, (short)0, c, false, false);
    const float bb = b3[laneM];
    #pragma unroll
    for (int r = 0; r < 8; ++r) {
      fbuf[(r + laneHi * 8) * 16 + laneM] = lrelu(c[r] + bb);
    }
  }
  __syncthreads();

  // ---- Phase 8: layer 4 — 16-wide dot per edge, store ----
  if (lane < 16) {
    float acc = b4[0];
    const float* rowf = fbuf + lane * 16;
    #pragma unroll
    for (int c = 0; c < 16; ++c) acc += rowf[c] * W4[c];
    const int e = e0 + lane;
    if (e < E) out[e] = acc;
  }
}

// ---------------------------------------------------------------------------
extern "C" void kernel_launch(void* const* d_in, const int* in_sizes, int n_in,
                              void* d_out, int out_size, void* d_ws, size_t ws_size,
                              hipStream_t stream) {
  const float* xd  = (const float*)d_in[0];   // x_drug    [10000,100] f32
  const float* xs  = (const float*)d_in[1];   // x_disease [10000,100] f32
  const int*   eli = (const int*)  d_in[2];   // edge_label_index [2,E] int32
  const float* W1  = (const float*)d_in[3];
  const float* b1  = (const float*)d_in[4];
  const float* W2  = (const float*)d_in[5];
  const float* b2  = (const float*)d_in[6];
  const float* W3  = (const float*)d_in[7];
  const float* b3  = (const float*)d_in[8];
  const float* W4  = (const float*)d_in[9];
  const float* b4  = (const float*)d_in[10];
  float* out = (float*)d_out;

  const int E = in_sizes[2] / 2;

  // workspace layout: packed f16 weight fragments
  _Float16* w1p = (_Float16*)d_ws;      // 56 frags * 512 halfs = 57,344 B
  _Float16* w2p = w1p + 56 * 512;       //  8 frags             =  8,192 B
  _Float16* w3p = w2p + 8 * 512;        //  1 frag              =  1,024 B

  pack_weights<<<65, 256, 0, stream>>>(W1, W2, W3, w1p, w2p, w3p);

  const int blocks = (E + EDGES_PER_BLOCK - 1) / EDGES_PER_BLOCK;
  mlp_edge_kernel<<<blocks, 256, 0, stream>>>(xd, xs, eli, w1p, w2p, w3p,
                                              b1, b2, b3, W4, b4, out, E);
}
